// SynthesizerAttention_12232066859635
// MI455X (gfx1250) — compile-verified, matrix operand order checked by reading, not run on hardware
//
#include <hip/hip_runtime.h>
#include <hip/hip_bf16.h>

typedef __bf16 bf16_t;
typedef __attribute__((ext_vector_type(8)))  __bf16 v8bf;
typedef __attribute__((ext_vector_type(16))) __bf16 v16bf;
typedef __attribute__((ext_vector_type(8)))  float  v8f;

#define T_SEQ 1023
#define T_PAD 1024
#define C_DIM 1024
#define NHEAD 16
#define DK    64
#define BATCH 8
#define MROWS (BATCH * T_SEQ)   /* 8184 */

__device__ __forceinline__ int imin(int a, int b) { return a < b ? a : b; }

// 16-bit WMMA A/B fragment from K-contiguous memory (global or LDS).
// Caller passes p = row_ptr + k0 + kbase where kbase = 8*(lane>=16).
// elems 0..7 = p[0..7], elems 8..15 = p[16..23]  (CDNA5 ISA fragment layout).
// Two b128 loads + register concat; no element shuffling.
template <typename PT>
__device__ __forceinline__ v16bf load_frag_k(const PT* p) {
  v8bf lo = *reinterpret_cast<const v8bf*>(p);
  v8bf hi = *reinterpret_cast<const v8bf*>(p + 16);
  return __builtin_shufflevector(lo, hi, 0, 1, 2, 3, 4, 5, 6, 7,
                                 8, 9, 10, 11, 12, 13, 14, 15);
}

__device__ __forceinline__ v8f wmma_bf16(v16bf a, v16bf b, v8f c) {
  return __builtin_amdgcn_wmma_f32_16x16x32_bf16(false, a, false, b, (short)0, c,
                                                 false, false);
}

// ---------------------------------------------------------------------------
// Generic GEMM: out[m,n] = act( sum_k A[m,k]*W[n,k] + bias[n] )
// A: M x K bf16 row-major, W: N x K bf16 row-major.
// Block = 256 threads (8 waves); each wave computes a 32(M) x 64(N) tile,
// so each W fragment feeds two WMMAs.
// ---------------------------------------------------------------------------
template <bool RELU, bool OUT_BF16>
__global__ __launch_bounds__(256)
void gemm_bias_kernel(const bf16_t* __restrict__ A, const bf16_t* __restrict__ W,
                      const float* __restrict__ bias, bf16_t* __restrict__ outb,
                      float* __restrict__ outf, int M, int N, int K) {
  const int lane = threadIdx.x & 31;
  const int wave = threadIdx.x >> 5;
  const int half = lane >> 4;
  const int lm   = lane & 15;
  const int kb   = half * 8;
  const int rowbase = blockIdx.x * 256 + wave * 32;
  const int n0      = blockIdx.y * 64;

  const bf16_t* aptr0 = A + (size_t)imin(rowbase + lm,      M - 1) * K;
  const bf16_t* aptr1 = A + (size_t)imin(rowbase + 16 + lm, M - 1) * K;

  v8f acc[2][4] = {};
  for (int k0 = 0; k0 < K; k0 += 32) {
    __builtin_prefetch(aptr0 + k0 + 256, 0, 1);
    __builtin_prefetch(aptr1 + k0 + 256, 0, 1);
    v16bf a0 = load_frag_k(aptr0 + k0 + kb);
    v16bf a1 = load_frag_k(aptr1 + k0 + kb);
#pragma unroll
    for (int dn = 0; dn < 4; ++dn) {
      v16bf b = load_frag_k(W + (size_t)(n0 + dn * 16 + lm) * K + k0 + kb);
      acc[0][dn] = wmma_bf16(a0, b, acc[0][dn]);
      acc[1][dn] = wmma_bf16(a1, b, acc[1][dn]);
    }
  }

#pragma unroll
  for (int r = 0; r < 2; ++r)
#pragma unroll
    for (int dn = 0; dn < 4; ++dn) {
      const int col = n0 + dn * 16 + lm;
      const float bv = bias[col];
#pragma unroll
      for (int i = 0; i < 8; ++i) {
        const int row = rowbase + r * 16 + i + 8 * half;
        if (row < M) {
          float v = acc[r][dn][i] + bv;
          if (RELU) v = v > 0.f ? v : 0.f;
          if (OUT_BF16) outb[(size_t)row * N + col] = (bf16_t)v;
          else          outf[(size_t)row * N + col] = v;
        }
      }
    }
}

// ---------------------------------------------------------------------------
// Fused synthesizer attention: one wave per (batch, head, 16-query tile).
// scores = H1_tile(16 x 64) @ w2t^T + b2; causal mask; online softmax;
// O += P(16 x 32) @ V(32 x 64).  All operand fragments are direct b128 loads
// thanks to N-major pre-transposed w2t / vt layouts.
// ---------------------------------------------------------------------------
__global__ __launch_bounds__(32)
void attn_kernel(const bf16_t* __restrict__ H1, const bf16_t* __restrict__ VT,
                 const bf16_t* __restrict__ W2T, const float* __restrict__ b2,
                 bf16_t* __restrict__ Y) {
  __shared__ bf16_t sP[16 * 32];   // prob tile: C-layout -> A-layout bounce

  const int lane = threadIdx.x;
  const int half = lane >> 4, lm = lane & 15, kb = half * 8;
  const int qt = blockIdx.x, h = blockIdx.y, b = blockIdx.z;
  const int qbase = qt * 16;

  // A fragments of relu(H1) for this query tile, K = d_k = 64 (two K=32 frags)
  const int qrow = imin(qbase + lm, T_SEQ - 1);
  const bf16_t* arow = H1 + (size_t)(b * T_SEQ + qrow) * C_DIM + h * DK;
  const v16bf aq0 = load_frag_k(arow + kb);
  const v16bf aq1 = load_frag_k(arow + 32 + kb);

  // Per-head transposed V: vt row (d) = VT + ((b*H + h)*DK + d) * T_PAD
  const bf16_t* vhead = VT + (size_t)((b * NHEAD + h) * DK) * T_PAD;

  v8f O[4] = {};
  float m_run[8], l_run[8];
#pragma unroll
  for (int i = 0; i < 8; ++i) { m_run[i] = -__builtin_inff(); l_run[i] = 0.f; }

  const int qmax = imin(qbase + 15, T_SEQ - 1);
  const int nsteps = qmax / 32 + 1;

  for (int st = 0; st < nsteps; ++st) {
    const int s0 = st * 32;

    // V B-fragments (K = 32 keys, N = 64 d): direct b128 loads from vt.
    v16bf bv[4];
#pragma unroll
    for (int dn = 0; dn < 4; ++dn)
      bv[dn] = load_frag_k(vhead + (size_t)(dn * 16 + lm) * T_PAD + s0 + kb);

    // Scores: 16 rows x 32 keys = two 16x16 C frags, K = d_k = 64 chained.
    // B frags direct from key-major w2t (row = key, 64 k's contiguous).
    v8f s[2];
#pragma unroll
    for (int t = 0; t < 2; ++t) {
      const bf16_t* wrow = W2T + (size_t)(s0 + t * 16 + lm) * DK;
      v8f z = {};
      z    = wmma_bf16(aq0, load_frag_k(wrow + kb), z);
      s[t] = wmma_bf16(aq1, load_frag_k(wrow + 32 + kb), z);
    }

    // Bias + causal mask + online softmax (row stats across 16-lane groups).
    const float bias0 = b2[imin(s0 + lm, T_SEQ - 1)];
    const float bias1 = b2[imin(s0 + 16 + lm, T_SEQ - 1)];
    float scl[8];
#pragma unroll
    for (int i = 0; i < 8; ++i) {
      const int q = qbase + i + 8 * half;
      const int key0 = s0 + lm, key1 = s0 + 16 + lm;
      float v0 = s[0][i] + bias0;
      float v1 = s[1][i] + bias1;
      if (key0 > q || key0 >= T_SEQ || q >= T_SEQ) v0 = -1e10f;
      if (key1 > q || key1 >= T_SEQ || q >= T_SEQ) v1 = -1e10f;
      float mx = fmaxf(v0, v1);
      mx = fmaxf(mx, __shfl_xor(mx, 1));
      mx = fmaxf(mx, __shfl_xor(mx, 2));
      mx = fmaxf(mx, __shfl_xor(mx, 4));
      mx = fmaxf(mx, __shfl_xor(mx, 8));
      const float mnew = fmaxf(m_run[i], mx);
      const float sc = __expf(m_run[i] - mnew);
      m_run[i] = mnew;
      const float p0 = __expf(v0 - mnew);
      const float p1 = __expf(v1 - mnew);
      float ps = p0 + p1;
      ps += __shfl_xor(ps, 1);
      ps += __shfl_xor(ps, 2);
      ps += __shfl_xor(ps, 4);
      ps += __shfl_xor(ps, 8);
      l_run[i] = l_run[i] * sc + ps;
      scl[i] = sc;
      s[0][i] = p0;
      s[1][i] = p1;
    }
#pragma unroll
    for (int dn = 0; dn < 4; ++dn)
#pragma unroll
      for (int i = 0; i < 8; ++i) O[dn][i] *= scl[i];

    // Transpose P from C-layout to A-layout via LDS bounce; read-back is
    // two ds_load_b128 per lane.
#pragma unroll
    for (int i = 0; i < 8; ++i) {
      sP[(i + 8 * half) * 32 + lm]      = (bf16_t)s[0][i];
      sP[(i + 8 * half) * 32 + 16 + lm] = (bf16_t)s[1][i];
    }
    __syncthreads();
    const v16bf ap = load_frag_k(sP + lm * 32 + kb);
#pragma unroll
    for (int dn = 0; dn < 4; ++dn) O[dn] = wmma_bf16(ap, bv[dn], O[dn]);
    __syncthreads();
  }

  // Normalize and store Y in bf16 (consumed by final GEMM).
#pragma unroll
  for (int dn = 0; dn < 4; ++dn)
#pragma unroll
    for (int i = 0; i < 8; ++i) {
      const int q = qbase + i + 8 * half;
      if (q < T_SEQ)
        Y[(size_t)(b * T_SEQ + q) * C_DIM + h * DK + dn * 16 + lm] =
            (bf16_t)(O[dn][i] / l_run[i]);
    }
}

// ---------------------------------------------------------------------------
// Preprocessing
// ---------------------------------------------------------------------------
__global__ void f32_to_bf16_kernel(const float* __restrict__ in,
                                   bf16_t* __restrict__ out, long long n) {
  const long long i = (long long)blockIdx.x * blockDim.x + threadIdx.x;
  if (i < n) out[i] = (bf16_t)in[i];
}

// w2 (d_k x T) f32 -> key-major bf16 w2t[key][d_k], key padded to T_PAD.
__global__ void w2_transpose_kernel(const float* __restrict__ w2,
                                    bf16_t* __restrict__ w2t) {
  const int i = blockIdx.x * blockDim.x + threadIdx.x;  // T_PAD*DK threads
  const int k = i & (DK - 1), key = i >> 6;
  w2t[i] = (bf16_t)((key < T_SEQ) ? w2[k * T_SEQ + key] : 0.f);
}

// V (B*T x C) bf16 -> per-head d-major vt[((b*H+h)*DK + d)][t], t padded.
__global__ void v_transpose_kernel(const bf16_t* __restrict__ v,
                                   bf16_t* __restrict__ vt) {
  const int i = blockIdx.x * blockDim.x + threadIdx.x;  // B*H*DK*T_PAD threads
  const int t = i & (T_PAD - 1);
  const int d = (i >> 10) & (DK - 1);
  const int h = (i >> 16) & (NHEAD - 1);
  const int b = i >> 20;
  bf16_t val = (bf16_t)0.f;
  if (t < T_SEQ) val = v[(size_t)(b * T_SEQ + t) * C_DIM + h * DK + d];
  vt[i] = val;
}

// ---------------------------------------------------------------------------
extern "C" void kernel_launch(void* const* d_in, const int* in_sizes, int n_in,
                              void* d_out, int out_size, void* d_ws, size_t ws_size,
                              hipStream_t stream) {
  const float* x   = (const float*)d_in[0];
  const float* w1w = (const float*)d_in[1];
  const float* w1b = (const float*)d_in[2];
  const float* w2  = (const float*)d_in[3];
  const float* b2  = (const float*)d_in[4];
  const float* vw  = (const float*)d_in[5];
  const float* vb  = (const float*)d_in[6];
  const float* pw  = (const float*)d_in[7];
  const float* pb  = (const float*)d_in[8];
  float* out = (float*)d_out;

  char* ws = (char*)d_ws;
  size_t off = 0;
  auto alloc = [&](size_t bytes) -> char* {
    char* p = ws + off;
    off = (off + bytes + 255) & ~(size_t)255;
    return p;
  };
  bf16_t* xb  = (bf16_t*)alloc((size_t)MROWS * C_DIM * sizeof(bf16_t));
  bf16_t* w1c = (bf16_t*)alloc((size_t)C_DIM * C_DIM * sizeof(bf16_t));
  bf16_t* vwc = (bf16_t*)alloc((size_t)C_DIM * C_DIM * sizeof(bf16_t));
  bf16_t* pwc = (bf16_t*)alloc((size_t)C_DIM * C_DIM * sizeof(bf16_t));
  bf16_t* w2t = (bf16_t*)alloc((size_t)T_PAD * DK * sizeof(bf16_t));
  bf16_t* h1b = (bf16_t*)alloc((size_t)MROWS * C_DIM * sizeof(bf16_t));
  bf16_t* vbf = (bf16_t*)alloc((size_t)MROWS * C_DIM * sizeof(bf16_t));
  bf16_t* vt  = (bf16_t*)alloc((size_t)BATCH * NHEAD * DK * T_PAD * sizeof(bf16_t));
  bf16_t* yb  = (bf16_t*)alloc((size_t)MROWS * C_DIM * sizeof(bf16_t));

  const long long nx = (long long)MROWS * C_DIM;
  const long long nw = (long long)C_DIM * C_DIM;
  f32_to_bf16_kernel<<<(unsigned)((nx + 255) / 256), 256, 0, stream>>>(x, xb, nx);
  f32_to_bf16_kernel<<<(unsigned)((nw + 255) / 256), 256, 0, stream>>>(w1w, w1c, nw);
  f32_to_bf16_kernel<<<(unsigned)((nw + 255) / 256), 256, 0, stream>>>(vw, vwc, nw);
  f32_to_bf16_kernel<<<(unsigned)((nw + 255) / 256), 256, 0, stream>>>(pw, pwc, nw);
  w2_transpose_kernel<<<(T_PAD * DK) / 256, 256, 0, stream>>>(w2, w2t);

  const dim3 ggrid((MROWS + 255) / 256, C_DIM / 64);
  // H1 = relu(x @ w1^T + b1)
  gemm_bias_kernel<true, true><<<ggrid, 256, 0, stream>>>(
      xb, w1c, w1b, h1b, nullptr, MROWS, C_DIM, C_DIM);
  // V = x @ value_w^T + vb
  gemm_bias_kernel<false, true><<<ggrid, 256, 0, stream>>>(
      xb, vwc, vb, vbf, nullptr, MROWS, C_DIM, C_DIM);
  // V -> per-head transposed layout for direct B-fragment loads
  v_transpose_kernel<<<(BATCH * NHEAD * DK * T_PAD) / 256, 256, 0, stream>>>(vbf, vt);

  // Fused scores/softmax/P@V
  const dim3 agrid((T_SEQ + 15) / 16, NHEAD, BATCH);
  attn_kernel<<<agrid, 32, 0, stream>>>(h1b, vt, w2t, b2, yb);

  // out = Y @ proj_w^T + pb  (f32 output)
  gemm_bias_kernel<false, false><<<ggrid, 256, 0, stream>>>(
      yb, pwc, pb, nullptr, out, MROWS, C_DIM, C_DIM);
}
